// RadiusVectorField_82162724373125
// MI455X (gfx1250) — matrix operand order repaired and checked by reading, not compiled, not run on hardware
//
#include <hip/hip_runtime.h>

// ---------------------------------------------------------------------------
// RadiusVectorField for MI455X (gfx1250, wave32, WMMA)
// ---------------------------------------------------------------------------

typedef __attribute__((ext_vector_type(16))) _Float16 v16h;
typedef __attribute__((ext_vector_type(8)))  float    v8f;

#define NQTOT 8192          // B * NQ
#define NCTX  8192          // context points per batch
#define HCH   64
#define KNN   32
#define R2F   0.04f
#define RF    0.2f
#define PI_OVER_R 15.707963267948966f   // pi / 0.2
#define CAND_CAP 1024

// ---- workspace layout (bytes) ----
#define WS_W1F   0                      // 16384 f16  (layer1 B-fragments)
#define WS_W2F   32768                  // 16384 f16
#define WS_W3F   65536                  // 32768 f16
#define WS_BN    131072                 // 1024 f32: s1,t1,s2,t2,s3(256),t3(256)
#define WS_IDX   135168                 // 8192*32 int
#define WS_DIST  (135168 + 1048576)     // 8192*32 f32
#define WS_Y     (135168 + 2097152)     // 8192*256 f32
// total = 10,620,928 bytes

// ===========================================================================
// Fragment helpers (layouts per CDNA5 ISA, 16-bit A/B, f32 C/D)
// ===========================================================================
union Frag16 { uint4 q[2]; v16h h; };

__device__ __forceinline__ v8f zero8() {
  v8f z;
#pragma unroll
  for (int i = 0; i < 8; ++i) z[i] = 0.0f;
  return z;
}

// A (16x32 f16): lane L holds row M=L%16; K-chunks [g*8, g*8+8) and [g*8+16, g*8+24), g=L/16
__device__ __forceinline__ v16h ld_fragA(const _Float16* X, int mt, int kt, int lane) {
  const int row = mt * 16 + (lane & 15);
  const int g   = lane >> 4;
  const char* p = (const char*)X + row * 256 + kt * 64 + g * 16;
  Frag16 f;
  f.q[0] = *(const uint4*)(p);
  f.q[1] = *(const uint4*)(p + 32);
  return f.h;
}

// B fragments are pre-swizzled by k_prep: frag stream of 32 lanes x 16 f16 (32B contiguous/lane)
__device__ __forceinline__ v16h ld_fragB(const _Float16* W, int frag, int lane) {
  const char* p = (const char*)W + ((size_t)frag * 512 + (size_t)lane * 16) * 2;
  Frag16 f;
  f.q[0] = *(const uint4*)(p);
  f.q[1] = *(const uint4*)(p + 16);
  return f.h;
}

// ===========================================================================
// k_prep: fold BN, convert weights to f16 in WMMA B-fragment order.
//   N = nt*16 + lane%16 ; K = kt*32 + (lane/16)*8 + (s%8) + (s/8)*16, s=0..15 packed per lane.
// ===========================================================================
__device__ __forceinline__ _Float16 frag_elem(const float* W, int e, int NT, int Kdim) {
  const int frag = e >> 9;
  const int sub  = e & 511;
  const int lane = sub >> 4;
  const int s16  = sub & 15;
  const int kt = frag / NT;
  const int nt = frag - kt * NT;
  const int K = kt * 32 + (lane >> 4) * 8 + (s16 & 7) + (s16 >> 3) * 16;
  const int N = nt * 16 + (lane & 15);
  return (_Float16)W[N * Kdim + K];
}

__global__ void k_prep(const float* __restrict__ w1, const float* __restrict__ w2,
                       const float* __restrict__ w3,
                       const float* __restrict__ b1, const float* __restrict__ g1,
                       const float* __restrict__ be1, const float* __restrict__ m1,
                       const float* __restrict__ v1,
                       const float* __restrict__ b2, const float* __restrict__ g2,
                       const float* __restrict__ be2, const float* __restrict__ m2,
                       const float* __restrict__ v2,
                       const float* __restrict__ b3, const float* __restrict__ g3,
                       const float* __restrict__ be3, const float* __restrict__ m3,
                       const float* __restrict__ v3,
                       _Float16* __restrict__ w1f, _Float16* __restrict__ w2f,
                       _Float16* __restrict__ w3f, float* __restrict__ bn) {
  const int t = blockIdx.x * blockDim.x + threadIdx.x;
  if (t < 16384) {
    w1f[t] = frag_elem(w1, t, 8, 128);
  } else if (t < 32768) {
    w2f[t - 16384] = frag_elem(w2, t - 16384, 8, 128);
  } else if (t < 65536) {
    w3f[t - 32768] = frag_elem(w3, t - 32768, 16, 128);
  }
  if (t < 128) {
    const float s = g1[t] * rsqrtf(v1[t] + 1e-5f);
    bn[t] = s;
    bn[128 + t] = s * (b1[t] - m1[t]) + be1[t];
  } else if (t < 256) {
    const int j = t - 128;
    const float s = g2[j] * rsqrtf(v2[j] + 1e-5f);
    bn[256 + j] = s;
    bn[384 + j] = s * (b2[j] - m2[j]) + be2[j];
  } else if (t < 512) {
    const int j = t - 256;
    const float s = g3[j] * rsqrtf(v3[j] + 1e-5f);
    bn[512 + j] = s;
    bn[768 + j] = s * (b3[j] - m3[j]) + be3[j];
  }
}

// ===========================================================================
// k_knn: one wave per query. Stage p_context (96KB) in LDS, radius-filter into
// per-wave LDS candidate lists (ds atomics), then 32 rounds of wave-wide
// min-extraction with shfl_xor butterflies. Output: 32 (idx,dist) per query,
// empty slots -> (0, 0) which get zero weight downstream (matches reference).
// ===========================================================================
#define KNN_WAVES 8
__global__ __launch_bounds__(256) void k_knn(const float* __restrict__ pq_all,
                                             const float* __restrict__ pc_all,
                                             int* __restrict__ idx_out,
                                             float* __restrict__ dist_out) {
  extern __shared__ __align__(16) char smem[];
  float*  ctx  = (float*)smem;                         // 8192*3 f32 = 98304B
  float2* cand = (float2*)(smem + 98304);              // 8 * 1024 * 8B = 65536B
  int*    cnts = (int*)(smem + 163840);                // 8 ints

  const int tid  = threadIdx.x;
  const int wave = tid >> 5;
  const int lane = tid & 31;
  const int q    = blockIdx.x * KNN_WAVES + wave;      // 4096 per batch, 8 | 4096
  const int b    = q >> 12;

  const float* pc = pc_all + (size_t)b * NCTX * 3;
  for (int i = tid; i < NCTX * 3; i += 256) ctx[i] = pc[i];
  if (tid < KNN_WAVES) cnts[tid] = 0;
  __syncthreads();

  const float qx = pq_all[q * 3 + 0];
  const float qy = pq_all[q * 3 + 1];
  const float qz = pq_all[q * 3 + 2];

  float2* my  = cand + wave * CAND_CAP;
  int*    myc = cnts + wave;

  for (int n = lane; n < NCTX; n += 32) {
    const float dx = ctx[n * 3 + 0] - qx;
    const float dy = ctx[n * 3 + 1] - qy;
    const float dz = ctx[n * 3 + 2] - qz;
    const float d = fmaf(dx, dx, fmaf(dy, dy, dz * dz));
    if (d <= R2F) {
      const int pos = atomicAdd(myc, 1);
      if (pos < CAND_CAP) my[pos] = make_float2(d, __int_as_float(n));
    }
  }
  asm volatile("" ::: "memory");   // DS pipe is in-order per wave; block reordering

  int M = *myc;
  M = M < CAND_CAP ? M : CAND_CAP;

  float outd = 0.0f;
  int   outi = 0;
  for (int r = 0; r < KNN; ++r) {
    float bv = 1e30f;
    int bi = 0, bp = -1;
    for (int e = lane; e < M; e += 32) {
      const float2 v = my[e];
      if (v.x < bv) { bv = v.x; bi = __float_as_int(v.y); bp = e; }
    }
#pragma unroll
    for (int off = 16; off > 0; off >>= 1) {
      const float ov = __shfl_xor(bv, off, 32);
      const int   oi = __shfl_xor(bi, off, 32);
      const int   op = __shfl_xor(bp, off, 32);
      if (ov < bv) { bv = ov; bi = oi; bp = op; }
    }
    if (bv < 1e29f) {
      if (lane == 0) my[bp].x = 1e30f;       // consume winner
      if (lane == r) { outd = bv; outi = bi; }
    }
    asm volatile("" ::: "memory");
  }
  idx_out[q * KNN + lane]  = outi;
  dist_out[q * KNN + lane] = outd;
}

// ===========================================================================
// k_group: WMMA grouped MLP. 8 waves/block, all weights in LDS, per-wave
// 32x128 f16 activation staging, 256 v_wmma_f32_16x16x32_f16 per query.
// B fragments are loaded ONCE per k-step and reused for both M-tiles
// (back-to-back WMMAs sharing SRC1) to halve LDS traffic in the hot loop.
// ===========================================================================
#define GW 8
// smem offsets (bytes)
#define SM_W2  32768
#define SM_W3  65536
#define SM_BN  131072
#define SM_XA  135168
#define SM_XB  200704
#define SM_WGT 266240
#define SM_WX  267264
#define SM_BX  268032
#define SM_TOT 268288

__device__ __forceinline__ void mlp_layer_relu(const _Float16* __restrict__ Xin,
                                               _Float16* __restrict__ Xout,
                                               const _Float16* __restrict__ Wf,
                                               const float* __restrict__ s,
                                               const float* __restrict__ t,
                                               int lane) {
  asm volatile("" ::: "memory");
  v8f acc[2][8];
#pragma unroll
  for (int mt = 0; mt < 2; ++mt)
#pragma unroll
    for (int i = 0; i < 8; ++i) acc[mt][i] = zero8();

#pragma unroll
  for (int kt = 0; kt < 4; ++kt) {
    const v16h a0 = ld_fragA(Xin, 0, kt, lane);
    const v16h a1 = ld_fragA(Xin, 1, kt, lane);
#pragma unroll
    for (int nt = 0; nt < 8; ++nt) {
      const v16h bfr = ld_fragB(Wf, kt * 8 + nt, lane);
      acc[0][nt] = __builtin_amdgcn_wmma_f32_16x16x32_f16(false, a0, false, bfr,
                                                          (short)0, acc[0][nt], false, false);
      acc[1][nt] = __builtin_amdgcn_wmma_f32_16x16x32_f16(false, a1, false, bfr,
                                                          (short)0, acc[1][nt], false, false);
    }
  }

  const int nl = lane & 15;
#pragma unroll
  for (int mt = 0; mt < 2; ++mt) {
    const int rowbase = mt * 16 + 8 * (lane >> 4);   // C/D: M = v + 8*(lane/16)
#pragma unroll
    for (int nt = 0; nt < 8; ++nt) {
      const int N = nt * 16 + nl;
      const float ss = s[N], tt = t[N];
#pragma unroll
      for (int v = 0; v < 8; ++v) {
        const float val = fmaxf(fmaf(acc[mt][nt][v], ss, tt), 0.0f);
        Xout[(rowbase + v) * 128 + N] = (_Float16)val;
      }
    }
  }
  asm volatile("" ::: "memory");
}

__global__ __launch_bounds__(256) void k_group(const float* __restrict__ pq,
                                               const float* __restrict__ pc,
                                               const float* __restrict__ hc,
                                               const float* __restrict__ wxyz,
                                               const float* __restrict__ bxyz,
                                               const void* __restrict__ wpack,
                                               const int* __restrict__ idx_in,
                                               const float* __restrict__ dist_in,
                                               float* __restrict__ y_out) {
  extern __shared__ __align__(16) char smem[];
  const _Float16* w1s = (const _Float16*)smem;
  const _Float16* w2s = (const _Float16*)(smem + SM_W2);
  const _Float16* w3s = (const _Float16*)(smem + SM_W3);
  const float*    bns = (const float*)(smem + SM_BN);
  float*          wgt = (float*)(smem + SM_WGT);
  float*          wxs = (float*)(smem + SM_WX);
  float*          bxs = (float*)(smem + SM_BX);

  const int tid  = threadIdx.x;
  const int wave = tid >> 5;
  const int lane = tid & 31;

  // weights + BN land in LDS exactly as laid out in ws (135168B contiguous)
  {
    uint* d = (uint*)smem;
    const uint* sct = (const uint*)wpack;
    for (int i = tid; i < 33792; i += 256) d[i] = sct[i];
  }
  for (int i = tid; i < 192; i += 256) wxs[i] = wxyz[i];
  if (tid < 64) bxs[tid] = bxyz[tid];
  __syncthreads();

  _Float16* Xa = (_Float16*)(smem + SM_XA) + wave * 4096;
  _Float16* Xb = (_Float16*)(smem + SM_XB) + wave * 4096;
  float*    W  = wgt + wave * 32;

  const int gw = blockIdx.x * GW + wave;
  const int stride = gridDim.x * GW;

  for (int q = gw; q < NQTOT; q += stride) {
    const int b = q >> 12;

    // ---- build input tile: lane = neighbor row -------------------------
    const int   nid = idx_in[q * KNN + lane];
    const float dd  = dist_in[q * KNN + lane];
    float w = 0.0f;
    if (dd > 0.0f && dd <= RF) w = 0.5f * (__cosf(dd * PI_OVER_R) + 1.0f);
    W[lane] = w;

    const float qx = pq[q * 3 + 0], qy = pq[q * 3 + 1], qz = pq[q * 3 + 2];
    const float* cp = pc + ((size_t)b * NCTX + nid) * 3;
    const float rx = cp[0] - qx, ry = cp[1] - qy, rz = cp[2] - qz;

#pragma unroll 8
    for (int j = 0; j < 64; ++j) {
      const float hv = fmaf(rx, wxs[j * 3 + 0],
                       fmaf(ry, wxs[j * 3 + 1],
                       fmaf(rz, wxs[j * 3 + 2], bxs[j])));
      Xa[lane * 128 + j] = (_Float16)hv;
    }
    const float4* hp = (const float4*)(hc + ((size_t)b * NCTX + nid) * HCH);
#pragma unroll
    for (int i = 0; i < 16; ++i) {
      const float4 h4 = hp[i];
      union { _Float16 h[4]; uint2 u; } pk;
      pk.h[0] = (_Float16)h4.x; pk.h[1] = (_Float16)h4.y;
      pk.h[2] = (_Float16)h4.z; pk.h[3] = (_Float16)h4.w;
      *(uint2*)&Xa[lane * 128 + 64 + i * 4] = pk.u;
    }

    // ---- layers 1 & 2 (relu) -------------------------------------------
    mlp_layer_relu(Xa, Xb, w1s, bns + 0,   bns + 128, lane);
    mlp_layer_relu(Xb, Xa, w2s, bns + 256, bns + 384, lane);

    // ---- layer 3 (no relu) + cosine-weighted row sum -------------------
    asm volatile("" ::: "memory");
    const float* s3 = bns + 512;
    const float* t3 = bns + 768;
    const int nl = lane & 15;
#pragma unroll
    for (int half = 0; half < 2; ++half) {
      v8f acc[2][8];
#pragma unroll
      for (int mt = 0; mt < 2; ++mt)
#pragma unroll
        for (int i = 0; i < 8; ++i) acc[mt][i] = zero8();

#pragma unroll
      for (int kt = 0; kt < 4; ++kt) {
        const v16h a0 = ld_fragA(Xa, 0, kt, lane);
        const v16h a1 = ld_fragA(Xa, 1, kt, lane);
#pragma unroll
        for (int nt = 0; nt < 8; ++nt) {
          const v16h bfr = ld_fragB(w3s, kt * 16 + half * 8 + nt, lane);
          acc[0][nt] = __builtin_amdgcn_wmma_f32_16x16x32_f16(false, a0, false, bfr,
                                                              (short)0, acc[0][nt], false, false);
          acc[1][nt] = __builtin_amdgcn_wmma_f32_16x16x32_f16(false, a1, false, bfr,
                                                              (short)0, acc[1][nt], false, false);
        }
      }
#pragma unroll
      for (int nt = 0; nt < 8; ++nt) {
        const int N = half * 128 + nt * 16 + nl;
        const float ss = s3[N], tt = t3[N];
        float p = 0.0f;
#pragma unroll
        for (int mt = 0; mt < 2; ++mt) {
          const int rowbase = mt * 16 + 8 * (lane >> 4);
#pragma unroll
          for (int v = 0; v < 8; ++v)
            p = fmaf(W[rowbase + v], fmaf(acc[mt][nt][v], ss, tt), p);
        }
        const float o = p + __shfl_xor(p, 16, 32);
        if (lane < 16)
          y_out[(size_t)q * 256 + half * 128 + nt * 16 + lane] = o;
      }
    }
    asm volatile("" ::: "memory");
  }
}

// ===========================================================================
// k_head: per-query 256->128->64->3 with LDS-cached f32 weights.
// ===========================================================================
__global__ __launch_bounds__(128) void k_head(const float* __restrict__ y_in,
                                              const float* __restrict__ w1,
                                              const float* __restrict__ b1,
                                              const float* __restrict__ g1,
                                              const float* __restrict__ be1,
                                              const float* __restrict__ m1,
                                              const float* __restrict__ v1,
                                              const float* __restrict__ w2,
                                              const float* __restrict__ b2,
                                              const float* __restrict__ g2,
                                              const float* __restrict__ be2,
                                              const float* __restrict__ m2,
                                              const float* __restrict__ v2,
                                              const float* __restrict__ w3,
                                              const float* __restrict__ b3,
                                              float* __restrict__ out) {
  extern __shared__ __align__(16) char smem[];
  float* w1s = (float*)smem;                // 128*256
  float* w2s = w1s + 32768;                 // 64*128
  float* w3s = w2s + 8192;                  // 3*64
  float* yb  = w3s + 192;                   // 256
  float* h1  = yb + 256;                    // 128
  float* h2  = h1 + 128;                    // 64

  const int tid = threadIdx.x;
  for (int i = tid; i < 32768; i += 128) w1s[i] = w1[i];
  for (int i = tid; i < 8192; i += 128) w2s[i] = w2[i];
  if (tid < 128) { // 192 elems, two steps
    w3s[tid] = w3[tid];
    if (tid < 64) w3s[128 + tid] = w3[128 + tid];
  }
  float s1 = 0.f, t1 = 0.f, s2 = 0.f, t2 = 0.f;
  if (tid < 128) { s1 = g1[tid] * rsqrtf(v1[tid] + 1e-5f); t1 = s1 * (b1[tid] - m1[tid]) + be1[tid]; }
  if (tid < 64)  { s2 = g2[tid] * rsqrtf(v2[tid] + 1e-5f); t2 = s2 * (b2[tid] - m2[tid]) + be2[tid]; }
  __syncthreads();

  for (int q = blockIdx.x; q < NQTOT; q += gridDim.x) {
    for (int i = tid; i < 256; i += 128) yb[i] = y_in[(size_t)q * 256 + i];
    __syncthreads();
    if (tid < 128) {
      float a = 0.0f;
      for (int k = 0; k < 256; ++k) a = fmaf(yb[k], w1s[tid * 256 + k], a);
      h1[tid] = fmaxf(fmaf(a, s1, t1), 0.0f);
    }
    __syncthreads();
    if (tid < 64) {
      float a = 0.0f;
      for (int k = 0; k < 128; ++k) a = fmaf(h1[k], w2s[tid * 128 + k], a);
      h2[tid] = fmaxf(fmaf(a, s2, t2), 0.0f);
    }
    __syncthreads();
    if (tid < 3) {
      float a = b3[tid];
      for (int k = 0; k < 64; ++k) a = fmaf(h2[k], w3s[tid * 64 + k], a);
      out[q * 3 + tid] = a;
    }
    __syncthreads();
  }
}

// ===========================================================================
// launcher
// ===========================================================================
extern "C" void kernel_launch(void* const* d_in, const int* in_sizes, int n_in,
                              void* d_out, int out_size, void* d_ws, size_t ws_size,
                              hipStream_t stream) {
  const float* p_query   = (const float*)d_in[0];
  const float* p_context = (const float*)d_in[1];
  const float* h_context = (const float*)d_in[2];
  const float* w_xyz     = (const float*)d_in[3];
  const float* b_xyz     = (const float*)d_in[4];
  const float* pw_w1 = (const float*)d_in[5];
  const float* pw_b1 = (const float*)d_in[6];
  const float* pw_g1 = (const float*)d_in[7];
  const float* pw_be1 = (const float*)d_in[8];
  const float* pw_m1 = (const float*)d_in[9];
  const float* pw_v1 = (const float*)d_in[10];
  const float* pw_w2 = (const float*)d_in[11];
  const float* pw_b2 = (const float*)d_in[12];
  const float* pw_g2 = (const float*)d_in[13];
  const float* pw_be2 = (const float*)d_in[14];
  const float* pw_m2 = (const float*)d_in[15];
  const float* pw_v2 = (const float*)d_in[16];
  const float* pw_w3 = (const float*)d_in[17];
  const float* pw_b3 = (const float*)d_in[18];
  const float* pw_g3 = (const float*)d_in[19];
  const float* pw_be3 = (const float*)d_in[20];
  const float* pw_m3 = (const float*)d_in[21];
  const float* pw_v3 = (const float*)d_in[22];
  const float* g_w1 = (const float*)d_in[23];
  const float* g_b1 = (const float*)d_in[24];
  const float* g_g1 = (const float*)d_in[25];
  const float* g_be1 = (const float*)d_in[26];
  const float* g_m1 = (const float*)d_in[27];
  const float* g_v1 = (const float*)d_in[28];
  const float* g_w2 = (const float*)d_in[29];
  const float* g_b2 = (const float*)d_in[30];
  const float* g_g2 = (const float*)d_in[31];
  const float* g_be2 = (const float*)d_in[32];
  const float* g_m2 = (const float*)d_in[33];
  const float* g_v2 = (const float*)d_in[34];
  const float* g_w3 = (const float*)d_in[35];
  const float* g_b3 = (const float*)d_in[36];
  (void)in_sizes; (void)n_in; (void)out_size; (void)ws_size;

  char* ws = (char*)d_ws;
  _Float16* w1f  = (_Float16*)(ws + WS_W1F);
  _Float16* w2f  = (_Float16*)(ws + WS_W2F);
  _Float16* w3f  = (_Float16*)(ws + WS_W3F);
  float*    bn   = (float*)(ws + WS_BN);
  int*      idxw = (int*)(ws + WS_IDX);
  float*    dstw = (float*)(ws + WS_DIST);
  float*    yw   = (float*)(ws + WS_Y);

  k_prep<<<256, 256, 0, stream>>>(pw_w1, pw_w2, pw_w3,
                                  pw_b1, pw_g1, pw_be1, pw_m1, pw_v1,
                                  pw_b2, pw_g2, pw_be2, pw_m2, pw_v2,
                                  pw_b3, pw_g3, pw_be3, pw_m3, pw_v3,
                                  w1f, w2f, w3f, bn);

  k_knn<<<NQTOT / KNN_WAVES, 256, 163840 + 64, stream>>>(p_query, p_context,
                                                         idxw, dstw);

  k_group<<<256, 256, SM_TOT, stream>>>(p_query, p_context, h_context,
                                        w_xyz, b_xyz, (const void*)w1f,
                                        idxw, dstw, yw);

  k_head<<<128, 128, (32768 + 8192 + 192 + 256 + 128 + 64) * 4, stream>>>(
      yw, g_w1, g_b1, g_g1, g_be1, g_m1, g_v1,
      g_w2, g_b2, g_g2, g_be2, g_m2, g_v2, g_w3, g_b3, (float*)d_out);
}
